// MambaBlock_83897891160068
// MI455X (gfx1250) — compile-verified, matrix operand order checked by reading, not run on hardware
//
#include <hip/hip_runtime.h>
#include <hip/hip_bf16.h>
#include <math.h>

typedef _Float16 half_t;
typedef __attribute__((ext_vector_type(16))) _Float16 v16h;
typedef __attribute__((ext_vector_type(8)))  _Float16 v8h;
typedef __attribute__((ext_vector_type(8)))  float    v8f;
typedef __attribute__((ext_vector_type(4)))  int      v4i;
typedef __attribute__((address_space(1))) v4i as1_v4i;
typedef __attribute__((address_space(3))) v4i as3_v4i;

#define BATCH   2
#define SEQ     2048
#define DMODEL  1024
#define DINNER  2048
#define N2      (2*DINNER)      // 4096
#define MROWS   (BATCH*SEQ)     // 4096
#define DSTATE  16
#define DCONV   4

// LDS tile: 128 rows x 32 halves, padded row stride 72 halves (144B).
// 144B is 16B-aligned (b128-legal) and 36-bank advance/row -> 16 distinct
// banks across the 16 lanes of a fragment read.
#define LDS_STRIDE 72
#define TILE_ELEMS (128 * LDS_STRIDE)

// ---------------------------------------------------------------------------
// 16B global->LDS copy on the gfx1250 async engine (ASYNCcnt, no VGPR staging).
// ---------------------------------------------------------------------------
__device__ __forceinline__ void cp16(half_t* lds_dst, const half_t* gsrc) {
#if defined(__has_builtin) && __has_builtin(__builtin_amdgcn_global_load_async_to_lds_b128)
    __builtin_amdgcn_global_load_async_to_lds_b128(
        (as1_v4i*)gsrc, (as3_v4i*)lds_dst, 0, 0);
#else
    *(v8h*)lds_dst = *(const v8h*)gsrc;
#endif
}

__device__ __forceinline__ void async_wait() {
#if defined(__has_builtin) && __has_builtin(__builtin_amdgcn_s_wait_asynccnt)
    __builtin_amdgcn_s_wait_asynccnt(0);
#else
    asm volatile("s_wait_asynccnt 0" ::: "memory");
#endif
}

// ---------------------------------------------------------------------------
// Cooperative fill: 256 threads, 128x32-half tile (512 x 16B chunks).
// ---------------------------------------------------------------------------
__device__ __forceinline__ void load_tile_g2l(half_t* __restrict__ lds,
                                              const half_t* __restrict__ gbase,
                                              int ld, int tid) {
    #pragma unroll
    for (int t = 0; t < 2; ++t) {
        int c   = tid + t * 256;        // 0..511
        int row = c >> 2;
        int seg = (c & 3) << 3;         // half offset: 0,8,16,24
        cp16(lds + row * LDS_STRIDE + seg, gbase + (size_t)row * ld + seg);
    }
}

// Causal-shift variant: tile row maps to global row (m - s); rows with
// l < s (before sequence start) are zero-filled.
__device__ __forceinline__ void load_tile_g2l_shift(half_t* __restrict__ lds,
                                                    const half_t* __restrict__ XS,
                                                    int lda, int row0, int k0,
                                                    int s, int tid) {
    #pragma unroll
    for (int t = 0; t < 2; ++t) {
        int c   = tid + t * 256;
        int row = c >> 2;
        int seg = (c & 3) << 3;
        int m   = row0 + row;
        int l   = m & (SEQ - 1);
        half_t* dst = lds + row * LDS_STRIDE + seg;
        if (l >= s) {
            cp16(dst, XS + (size_t)(m - s) * lda + k0 + seg);
        } else {
            v8h z = {};
            *(v8h*)dst = z;
        }
    }
}

// ---------------------------------------------------------------------------
// WMMA 16x32 fragment from an LDS tile (CDNA5 A/B-operand layout):
// lane<16: row r0+lane, K {0..7,16..23}; lane>=16: row r0+lane-16, K {8..15,24..31}
// ---------------------------------------------------------------------------
__device__ __forceinline__ v16h lds_frag(const half_t* base, int r0, int lane) {
    const half_t* q = base + (r0 + (lane & 15)) * LDS_STRIDE + ((lane >> 4) << 3);
    union { v16h v; v8h h[2]; } f;
    f.h[0] = *(const v8h*)q;
    f.h[1] = *(const v8h*)(q + 16);
    return f.v;
}

// Same fragment layout straight from global memory (for small GEMMs).
__device__ __forceinline__ v16h load_tile16x32(const half_t* __restrict__ p,
                                               int ld, int r0, int k0, int lane) {
    int r  = r0 + (lane & 15);
    int kb = k0 + ((lane >> 4) << 3);
    const half_t* q = p + (size_t)r * ld + kb;
    union { v16h v; v8h h[2]; } f;
    f.h[0] = *(const v8h*)q;
    f.h[1] = *(const v8h*)(q + 16);
    return f.v;
}

// ---------------------------------------------------------------------------
// Double-buffered tiled GEMM: C[M,N] = A[M,K]*Bt[N,K]^T + bias.
// 256 thr / 8 waves; workgroup tile 128x128; wave tile 64x32 (4x2 WMMA accs).
// Async fills of tile k+1 overlap WMMA on tile k; one asynccnt-wait + one
// barrier per k-step.
// ---------------------------------------------------------------------------
template <bool F16OUT>
__global__ __launch_bounds__(256)
void gemm_f16_tiled_kernel(const half_t* __restrict__ A, int lda,
                           const half_t* __restrict__ Bt, int ldb,
                           const float*  __restrict__ bias,
                           float* __restrict__ Cf, half_t* __restrict__ Ch,
                           int ldc, int K) {
    __shared__ half_t As[2][TILE_ELEMS];
    __shared__ half_t Bs[2][TILE_ELEMS];
    int tid  = threadIdx.x;
    int lane = tid & 31, wid = tid >> 5;
    int wr = wid & 1;          // wave row: 0..1  (64 rows each)
    int wc = wid >> 1;         // wave col: 0..3  (32 cols each)
    int row0 = blockIdx.y * 128, col0 = blockIdx.x * 128;
    const half_t* Abase = A  + (size_t)row0 * lda;
    const half_t* Bbase = Bt + (size_t)col0 * ldb;

    // prologue: fill buffer 0
    load_tile_g2l(As[0], Abase, lda, tid);
    load_tile_g2l(Bs[0], Bbase, ldb, tid);
    async_wait();
    __syncthreads();

    v8f acc[4][2] = {};
    int cur = 0;
    for (int k0 = 0; k0 < K; k0 += 32) {
        int nxt = cur ^ 1;
        if (k0 + 32 < K) {                 // async-prefetch next k-slab
            load_tile_g2l(As[nxt], Abase + k0 + 32, lda, tid);
            load_tile_g2l(Bs[nxt], Bbase + k0 + 32, ldb, tid);
        }
        v16h af[4], bf[2];
        #pragma unroll
        for (int i = 0; i < 4; ++i) af[i] = lds_frag(As[cur], wr * 64 + i * 16, lane);
        #pragma unroll
        for (int j = 0; j < 2; ++j) bf[j] = lds_frag(Bs[cur], wc * 32 + j * 16, lane);
        #pragma unroll
        for (int i = 0; i < 4; ++i)
            #pragma unroll
            for (int j = 0; j < 2; ++j)
                acc[i][j] = __builtin_amdgcn_wmma_f32_16x16x32_f16(
                    false, af[i], false, bf[j], (short)0, acc[i][j], false, false);
        async_wait();
        __syncthreads();
        cur = nxt;
    }

    int nlo = lane & 15, rhi = (lane >> 4) << 3;
    #pragma unroll
    for (int i = 0; i < 4; ++i)
        #pragma unroll
        for (int j = 0; j < 2; ++j) {
            int col = col0 + wc * 32 + j * 16 + nlo;
            float bv = bias[col];
            #pragma unroll
            for (int r = 0; r < 8; ++r) {
                int row = row0 + wr * 64 + i * 16 + rhi + r;
                float v = acc[i][j][r] + bv;
                size_t idx = (size_t)row * ldc + col;
                if (F16OUT) Ch[idx] = (half_t)v;
                else        Cf[idx] = v;
            }
        }
}

// ---------------------------------------------------------------------------
// Double-buffered tiled causal conv1d (full conv, k=4): 4 shift-accumulated
// GEMMs with a flattened (shift, k-slab) pipeline.
// Wt[k][o][i] = conv_w[o][i][k]  (Bt layout [N=o, K=i]).
// ---------------------------------------------------------------------------
__global__ __launch_bounds__(256)
void conv_gemm_tiled_kernel(const half_t* __restrict__ XS, int lda,
                            const half_t* __restrict__ Wt,
                            const float*  __restrict__ conv_b,
                            half_t* __restrict__ XCh) {
    __shared__ half_t As[2][TILE_ELEMS];
    __shared__ half_t Bs[2][TILE_ELEMS];
    int tid  = threadIdx.x;
    int lane = tid & 31, wid = tid >> 5;
    int wr = wid & 1, wc = wid >> 1;
    int row0 = blockIdx.y * 128, col0 = blockIdx.x * 128;

    const int KT_PER_SHIFT = DINNER / 32;               // 64
    const int KTILES = DCONV * KT_PER_SHIFT;            // 256

    auto fill = [&](int kt, int buf) {
        int kk = kt / KT_PER_SHIFT;
        int k0 = (kt % KT_PER_SHIFT) * 32;
        int s  = (DCONV - 1) - kk;
        load_tile_g2l_shift(As[buf], XS, lda, row0, k0, s, tid);
        load_tile_g2l(Bs[buf],
                      Wt + (size_t)kk * DINNER * DINNER + (size_t)col0 * DINNER + k0,
                      DINNER, tid);
    };

    fill(0, 0);
    async_wait();
    __syncthreads();

    v8f acc[4][2] = {};
    int cur = 0;
    for (int kt = 0; kt < KTILES; ++kt) {
        int nxt = cur ^ 1;
        if (kt + 1 < KTILES) fill(kt + 1, nxt);
        v16h af[4], bf[2];
        #pragma unroll
        for (int i = 0; i < 4; ++i) af[i] = lds_frag(As[cur], wr * 64 + i * 16, lane);
        #pragma unroll
        for (int j = 0; j < 2; ++j) bf[j] = lds_frag(Bs[cur], wc * 32 + j * 16, lane);
        #pragma unroll
        for (int i = 0; i < 4; ++i)
            #pragma unroll
            for (int j = 0; j < 2; ++j)
                acc[i][j] = __builtin_amdgcn_wmma_f32_16x16x32_f16(
                    false, af[i], false, bf[j], (short)0, acc[i][j], false, false);
        async_wait();
        __syncthreads();
        cur = nxt;
    }

    int nlo = lane & 15, rhi = (lane >> 4) << 3;
    #pragma unroll
    for (int i = 0; i < 4; ++i)
        #pragma unroll
        for (int j = 0; j < 2; ++j) {
            int col = col0 + wc * 32 + j * 16 + nlo;
            float bv = conv_b[col];
            #pragma unroll
            for (int r = 0; r < 8; ++r) {
                int row = row0 + wr * 64 + i * 16 + rhi + r;
                XCh[(size_t)row * DINNER + col] = (half_t)(acc[i][j][r] + bv);
            }
        }
}

// ---------------------------------------------------------------------------
// Small-N GEMM (one wave per 16x16 tile) — used for U = xc @ B_ssm (N=16).
// ---------------------------------------------------------------------------
__global__ void gemm_f16_small_kernel(const half_t* __restrict__ A, int lda,
                                      const half_t* __restrict__ Bt, int ldb,
                                      float* __restrict__ Cf, int ldc, int K) {
    int lane = threadIdx.x & 31;
    int col0 = blockIdx.x * 16;
    int row0 = blockIdx.y * 16;
    v8f c = {};
    for (int k0 = 0; k0 < K; k0 += 32) {
        v16h a = load_tile16x32(A,  lda, row0, k0, lane);
        v16h b = load_tile16x32(Bt, ldb, col0, k0, lane);
        c = __builtin_amdgcn_wmma_f32_16x16x32_f16(false, a, false, b,
                                                   (short)0, c, false, false);
    }
    int n  = col0 + (lane & 15);
    int rb = row0 + ((lane >> 4) << 3);
    #pragma unroll
    for (int r = 0; r < 8; ++r)
        Cf[(size_t)(rb + r) * ldc + n] = c[r];
}

// ---------------------------------------------------------------------------
// Sequential SSM scan: state' = state @ A + u_t. One wave per batch.
// ---------------------------------------------------------------------------
__global__ void ssm_scan_kernel(const float* __restrict__ U,
                                const float* __restrict__ Amat,
                                float* __restrict__ S,
                                half_t* __restrict__ Sh) {
    __shared__ float st[DSTATE];
    int b = blockIdx.x;
    int j = threadIdx.x;
    float acol[DSTATE];
    if (j < DSTATE) {
        #pragma unroll
        for (int i = 0; i < DSTATE; ++i) acol[i] = Amat[i * DSTATE + j];
        st[j] = 0.0f;
    }
    __syncthreads();
    for (int t = 0; t < SEQ; ++t) {
        size_t m = (size_t)b * SEQ + t;
        float acc = 0.0f;
        if (j < DSTATE) {
            acc = U[m * DSTATE + j];
            #pragma unroll
            for (int i = 0; i < DSTATE; ++i) acc += st[i] * acol[i];
        }
        __syncthreads();
        if (j < DSTATE) {
            st[j] = acc;
            S[m * DSTATE + j] = acc;
            Sh[m * 32 + j] = (half_t)acc;
            Sh[m * 32 + 16 + j] = (half_t)0;
        }
        __syncthreads();
    }
}

// ---------------------------------------------------------------------------
// Y = S @ C_ssm + xc * D_skip, gated by sigmoid(gate); f16 out for out_proj.
// ---------------------------------------------------------------------------
__global__ void ssm_y_gemm_kernel(const half_t* __restrict__ Sh,
                                  const half_t* __restrict__ Ct,
                                  const half_t* __restrict__ XCh,
                                  const float*  __restrict__ Dskip,
                                  const half_t* __restrict__ XPh,
                                  half_t* __restrict__ YG) {
    int lane = threadIdx.x & 31;
    int col0 = blockIdx.x * 16;
    int row0 = blockIdx.y * 16;
    v16h a = load_tile16x32(Sh, 32, row0, 0, lane);
    v16h b = load_tile16x32(Ct, 32, col0, 0, lane);
    v8f  c = {};
    c = __builtin_amdgcn_wmma_f32_16x16x32_f16(false, a, false, b,
                                               (short)0, c, false, false);
    int n  = col0 + (lane & 15);
    int rb = row0 + ((lane >> 4) << 3);
    float d = Dskip[n];
    #pragma unroll
    for (int r = 0; r < 8; ++r) {
        size_t row = (size_t)(rb + r);
        float xcv = (float)XCh[row * DINNER + n];
        float g   = (float)XPh[row * N2 + DINNER + n];
        float sig = 1.0f / (1.0f + __expf(-g));
        YG[row * DINNER + n] = (half_t)((c[r] + xcv * d) * sig);
    }
}

// ---------------------------------------------------------------------------
// Residual + LayerNorm (one block per row; OM already carries out_proj bias).
// ---------------------------------------------------------------------------
__global__ void ln_kernel(const float* __restrict__ OM,
                          const float* __restrict__ X,
                          const float* __restrict__ gamma,
                          const float* __restrict__ beta,
                          float* __restrict__ out) {
    __shared__ float red[256];
    int m = blockIdx.x, tid = threadIdx.x;
    float h[4]; float sum = 0.0f;
    #pragma unroll
    for (int i = 0; i < 4; ++i) {
        int jx = tid + i * 256;
        h[i] = OM[(size_t)m * DMODEL + jx] + X[(size_t)m * DMODEL + jx];
        sum += h[i];
    }
    red[tid] = sum; __syncthreads();
    for (int s = 128; s > 0; s >>= 1) {
        if (tid < s) red[tid] += red[tid + s];
        __syncthreads();
    }
    float mu = red[0] * (1.0f / DMODEL);
    __syncthreads();
    float vs = 0.0f;
    #pragma unroll
    for (int i = 0; i < 4; ++i) { float dd = h[i] - mu; vs += dd * dd; }
    red[tid] = vs; __syncthreads();
    for (int s = 128; s > 0; s >>= 1) {
        if (tid < s) red[tid] += red[tid + s];
        __syncthreads();
    }
    float inv = rsqrtf(red[0] * (1.0f / DMODEL) + 1e-5f);
    #pragma unroll
    for (int i = 0; i < 4; ++i) {
        int jx = tid + i * 256;
        out[(size_t)m * DMODEL + jx] = (h[i] - mu) * inv * gamma[jx] + beta[jx];
    }
}

__global__ void final_state_kernel(const float* __restrict__ S,
                                   float* __restrict__ dst) {
    int t = threadIdx.x;
    if (t < BATCH * DSTATE) {
        int b = t >> 4, n = t & 15;
        dst[t] = S[((size_t)b * SEQ + (SEQ - 1)) * DSTATE + n];
    }
}

// --------------------------- converters ------------------------------------
__global__ void f32_to_f16_kernel(const float* __restrict__ src,
                                  half_t* __restrict__ dst, size_t n) {
    size_t i = (size_t)blockIdx.x * blockDim.x + threadIdx.x;
    if (i < n) dst[i] = (half_t)src[i];
}

// dst[n*K + k] = src[k*N + n]
__global__ void transpose_to_f16_kernel(const float* __restrict__ src,
                                        half_t* __restrict__ dst,
                                        int K, int N) {
    size_t i = (size_t)blockIdx.x * blockDim.x + threadIdx.x;
    size_t total = (size_t)K * N;
    if (i < total) {
        size_t n = i / K, k = i % K;
        dst[i] = (half_t)src[k * (size_t)N + n];
    }
}

// dst[(k*C + o)*C + i] = conv_w[(o*C + i)*4 + k]
__global__ void conv_w_rearrange_kernel(const float* __restrict__ cw,
                                        half_t* __restrict__ dst) {
    size_t i = (size_t)blockIdx.x * blockDim.x + threadIdx.x;
    size_t total = (size_t)DCONV * DINNER * DINNER;
    if (i < total) {
        size_t k = i / ((size_t)DINNER * DINNER);
        size_t rem = i % ((size_t)DINNER * DINNER);
        size_t o = rem / DINNER, ii = rem % DINNER;
        dst[i] = (half_t)cw[(o * DINNER + ii) * DCONV + k];
    }
}

// dst[n*32 + k] = (k<16) ? C_ssm[k*DINNER + n] : 0
__global__ void csm_pad_transpose_kernel(const float* __restrict__ csm,
                                         half_t* __restrict__ dst) {
    size_t i = (size_t)blockIdx.x * blockDim.x + threadIdx.x;
    size_t total = (size_t)DINNER * 32;
    if (i < total) {
        size_t n = i / 32, k = i % 32;
        dst[i] = (k < DSTATE) ? (half_t)csm[k * (size_t)DINNER + n] : (half_t)0;
    }
}

// ---------------------------------------------------------------------------
extern "C" void kernel_launch(void* const* d_in, const int* in_sizes, int n_in,
                              void* d_out, int out_size, void* d_ws, size_t ws_size,
                              hipStream_t stream) {
    const float* x         = (const float*)d_in[0];
    const float* in_proj_w = (const float*)d_in[1];
    const float* in_proj_b = (const float*)d_in[2];
    const float* conv_w    = (const float*)d_in[3];
    const float* conv_b    = (const float*)d_in[4];
    const float* Amat      = (const float*)d_in[5];
    const float* B_ssm     = (const float*)d_in[6];
    const float* C_ssm     = (const float*)d_in[7];
    const float* D_skip    = (const float*)d_in[8];
    const float* out_proj_w= (const float*)d_in[9];
    const float* out_proj_b= (const float*)d_in[10];
    const float* ln_g      = (const float*)d_in[11];
    const float* ln_b      = (const float*)d_in[12];
    float* out = (float*)d_out;

    char* ws = (char*)d_ws;
    size_t off = 0;
    auto carve = [&](size_t bytes) -> char* {
        off = (off + 255) & ~(size_t)255;
        char* p = ws + off;
        off += bytes;
        return p;
    };
    half_t* xh      = (half_t*)carve((size_t)MROWS * DMODEL * 2);
    half_t* w_inT   = (half_t*)carve((size_t)N2 * DMODEL * 2);
    half_t* w_convT = (half_t*)carve((size_t)DCONV * DINNER * DINNER * 2);
    half_t* w_outT  = (half_t*)carve((size_t)DMODEL * DINNER * 2);
    half_t* bsmT    = (half_t*)carve((size_t)DSTATE * DINNER * 2);
    half_t* csmT    = (half_t*)carve((size_t)DINNER * 32 * 2);
    half_t* xp_h    = (half_t*)carve((size_t)MROWS * N2 * 2);
    half_t* xch     = (half_t*)carve((size_t)MROWS * DINNER * 2);
    float*  u       = (float*) carve((size_t)MROWS * DSTATE * 4);
    float*  s       = (float*) carve((size_t)MROWS * DSTATE * 4);
    half_t* sh      = (half_t*)carve((size_t)MROWS * 32 * 2);
    half_t* yg      = (half_t*)carve((size_t)MROWS * DINNER * 2);
    float*  om      = (float*) carve((size_t)MROWS * DMODEL * 4);
    (void)ws_size; (void)in_sizes; (void)n_in; (void)out_size;

    auto blocks = [](size_t n) { return (unsigned)((n + 255) / 256); };

    // 1) precision conversion / layout transforms
    f32_to_f16_kernel<<<blocks((size_t)MROWS*DMODEL), 256, 0, stream>>>(x, xh, (size_t)MROWS*DMODEL);
    transpose_to_f16_kernel<<<blocks((size_t)N2*DMODEL), 256, 0, stream>>>(in_proj_w, w_inT, DMODEL, N2);
    conv_w_rearrange_kernel<<<blocks((size_t)DCONV*DINNER*DINNER), 256, 0, stream>>>(conv_w, w_convT);
    transpose_to_f16_kernel<<<blocks((size_t)DMODEL*DINNER), 256, 0, stream>>>(out_proj_w, w_outT, DINNER, DMODEL);
    transpose_to_f16_kernel<<<blocks((size_t)DSTATE*DINNER), 256, 0, stream>>>(B_ssm, bsmT, DINNER, DSTATE);
    csm_pad_transpose_kernel<<<blocks((size_t)DINNER*32), 256, 0, stream>>>(C_ssm, csmT);

    // 2) in_proj: xp = x @ W_in + b   [4096 x 4096], f16 out (tiled, dbl-buffered)
    gemm_f16_tiled_kernel<true><<<dim3(N2/128, MROWS/128), 256, 0, stream>>>(
        xh, DMODEL, w_inT, DMODEL, in_proj_b, nullptr, xp_h, N2, DMODEL);

    // 3) causal conv: 4 shifted GEMMs fused (tiled, dbl-buffered)
    conv_gemm_tiled_kernel<<<dim3(DINNER/128, MROWS/128), 256, 0, stream>>>(
        xp_h, N2, w_convT, conv_b, xch);

    // 4) U = xc @ B_ssm   [4096 x 16], f32 out (small-N path)
    gemm_f16_small_kernel<<<dim3(DSTATE/16, MROWS/16), 32, 0, stream>>>(
        xch, DINNER, bsmT, DINNER, u, DSTATE, DINNER);

    // 5) sequential 16-state scan
    ssm_scan_kernel<<<BATCH, 32, 0, stream>>>(u, Amat, s, sh);

    // 6) Y = S @ C_ssm + xc*D, gated -> yg (f16)
    ssm_y_gemm_kernel<<<dim3(DINNER/16, MROWS/16), 32, 0, stream>>>(
        sh, csmT, xch, D_skip, xp_h, yg);

    // 7) out_proj: om = yg @ W_out + b   [4096 x 1024], f32 out (tiled, dbl-buffered)
    gemm_f16_tiled_kernel<false><<<dim3(DMODEL/128, MROWS/128), 256, 0, stream>>>(
        yg, DINNER, w_outT, DINNER, out_proj_b, om, nullptr, DMODEL, DINNER);

    // 8) residual + layernorm
    ln_kernel<<<MROWS, 256, 0, stream>>>(om, x, ln_g, ln_b, out);

    // 9) final_state
    final_state_kernel<<<1, 32, 0, stream>>>(s, out + (size_t)MROWS * DMODEL);
}
